// GAT_10514079941500
// MI455X (gfx1250) — compile-verified, matrix operand order checked by reading, not run on hardware
//
#include <hip/hip_runtime.h>

typedef __attribute__((ext_vector_type(16))) _Float16 v16h;
typedef __attribute__((ext_vector_type(8)))  _Float16 h8;
typedef __attribute__((ext_vector_type(2)))  _Float16 h2;
typedef __attribute__((ext_vector_type(2)))  __fp16   hp2;   // cvt_pkrtz result type
typedef __attribute__((ext_vector_type(8)))  float    v8f;
typedef __attribute__((ext_vector_type(4)))  float    v4f;

#define M_ROWS 6144
#define N_COLS 6144
#define K_DIM  512
#define LD     6145           // coupling matrix stride (6145 x 6145)
#define NTOT   (12288 * 512)

// workspace layout (float indices)
#define WS_PMIN   0           // 256 partial mins
#define WS_PMAX   256         // 256 partial maxs
#define WS_STATS  512         // [0]=min, [1]=1/(max-min)
#define WS_U      1024        // 6145 floats
#define WS_V      8192        // 6145 floats
#define NCHUNK    64          // row chunks for column-wise logsumexp
#define PSTRIDE   6400        // padded column stride for partials
#define WS_PM     16384       // NCHUNK*PSTRIDE partial maxes
#define WS_PS     (WS_PM + NCHUNK * PSTRIDE)  // partial sums

// ---------- online logsumexp combine: (m,s) <- (m,s) ⊕ (om,os) ----------
__device__ __forceinline__ void lse_combine(float& m, float& s, float om, float os) {
    if (om > m)      { s = s * __expf(m - om) + os; m = om; }
    else if (m > om) { s += os * __expf(om - m); }
    else             { s += os; }   // equal (also covers both == -inf)
}

// ---------------------- 1) min/max reduction ----------------------------
__global__ void k_minmax_partial(const float* __restrict__ x, float* __restrict__ ws) {
    int tid = threadIdx.x;
    int gid = blockIdx.x * blockDim.x + tid;
    int stride = gridDim.x * blockDim.x;
    float mn = 3.402823466e38f, mx = -3.402823466e38f;
    for (int i = gid; i < NTOT; i += stride) {
        float v = x[i];
        mn = fminf(mn, v);
        mx = fmaxf(mx, v);
    }
    for (int off = 16; off; off >>= 1) {
        mn = fminf(mn, __shfl_xor(mn, off, 32));
        mx = fmaxf(mx, __shfl_xor(mx, off, 32));
    }
    __shared__ float smn[8], smx[8];
    int wave = tid >> 5, lane = tid & 31;
    if (lane == 0) { smn[wave] = mn; smx[wave] = mx; }
    __syncthreads();
    if (tid == 0) {
        for (int w = 1; w < 8; ++w) { mn = fminf(mn, smn[w]); mx = fmaxf(mx, smx[w]); }
        ws[WS_PMIN + blockIdx.x] = mn;
        ws[WS_PMAX + blockIdx.x] = mx;
    }
}

__global__ void k_minmax_final(float* __restrict__ ws) {
    int tid = threadIdx.x;                 // 256 threads, 256 partials
    float mn = ws[WS_PMIN + tid];
    float mx = ws[WS_PMAX + tid];
    for (int off = 16; off; off >>= 1) {
        mn = fminf(mn, __shfl_xor(mn, off, 32));
        mx = fmaxf(mx, __shfl_xor(mx, off, 32));
    }
    __shared__ float smn[8], smx[8];
    int wave = tid >> 5, lane = tid & 31;
    if (lane == 0) { smn[wave] = mn; smx[wave] = mx; }
    __syncthreads();
    if (tid == 0) {
        for (int w = 1; w < 8; ++w) { mn = fminf(mn, smn[w]); mx = fmaxf(mx, smx[w]); }
        ws[WS_STATS]     = mn;
        ws[WS_STATS + 1] = 1.0f / (mx - mn);
    }
}

// pack 8 normalized floats (x*inv + bias) into 8 halfs via v_cvt_pk_rtz_f16_f32
__device__ __forceinline__ h8 pack8(const v4f& a, const v4f& b, float inv, float bias) {
    union { h8 v; hp2 p[4]; } u;
    u.p[0] = __builtin_amdgcn_cvt_pkrtz(fmaf(a[0], inv, bias), fmaf(a[1], inv, bias));
    u.p[1] = __builtin_amdgcn_cvt_pkrtz(fmaf(a[2], inv, bias), fmaf(a[3], inv, bias));
    u.p[2] = __builtin_amdgcn_cvt_pkrtz(fmaf(b[0], inv, bias), fmaf(b[1], inv, bias));
    u.p[3] = __builtin_amdgcn_cvt_pkrtz(fmaf(b[2], inv, bias), fmaf(b[3], inv, bias));
    return u.v;
}

// ------- 2) scores = norm(src) @ norm(tgt)^T via v_wmma_f32_16x16x32_f16 -------
// block tile 128x128, 8 waves (2x4), wave tile 64x32 = 4x2 WMMA 16x16 tiles.
__global__ __launch_bounds__(256) void k_gemm_wmma(const float* __restrict__ x,
                                                   const float* __restrict__ ws,
                                                   float* __restrict__ Cout) {
    const float mn   = ws[WS_STATS];
    const float inv  = ws[WS_STATS + 1];
    const float bias = -mn * inv;          // (v - mn) * inv == v*inv + bias

    __shared__ _Float16 As[128][40];   // +8 pad, row = 80B (16B-aligned rows)
    __shared__ _Float16 Bs[128][40];

    const int tid   = threadIdx.x;
    const int wave  = tid >> 5;
    const int lane  = tid & 31;
    const int wm    = wave >> 2;       // 0..1
    const int wn    = wave & 3;        // 0..3
    const int mrow  = lane & 15;       // M (or N) within 16x16 tile
    const int khalf = lane >> 4;       // K-half selector per ISA layout
    const int tileM = blockIdx.y * 128;
    const int tileN = blockIdx.x * 128;

    v8f acc[4][2] = {};

    for (int k0 = 0; k0 < K_DIM; k0 += 32) {
        __syncthreads();
        // cooperative staging: each thread owns 8 contiguous K elements
        // -> 2x global_load_b128 + packed cvt + 1x ds_store_b128 per chunk
        #pragma unroll
        for (int it = 0; it < 2; ++it) {
            int idx = it * 256 + tid;          // 512 chunks of 8 halfs = 128x32
            int r = idx >> 2;
            int c = (idx & 3) * 8;
            const float* pa = &x[(size_t)(tileM + r) * K_DIM + (k0 + c)];
            v4f a0 = *(const v4f*)pa;
            v4f a1 = *(const v4f*)(pa + 4);
            *(h8*)&As[r][c] = pack8(a0, a1, inv, bias);
            const float* pb = &x[(size_t)(M_ROWS + tileN + r) * K_DIM + (k0 + c)];
            v4f b0 = *(const v4f*)pb;
            v4f b1 = *(const v4f*)(pb + 4);
            *(h8*)&Bs[r][c] = pack8(b0, b1, inv, bias);
        }
        __syncthreads();

        // gather fragments per 16-bit A layout (ISA §7.12.2); B mirrored
        v16h afrag[4], bfrag[2];
        #pragma unroll
        for (int s = 0; s < 4; ++s) {
            int m = wm * 64 + s * 16 + mrow;
            #pragma unroll
            for (int j = 0; j < 8; ++j) {
                int kb = ((j < 4) ? (j * 2) : (16 + (j - 4) * 2)) + khalf * 8;
                h2 p = *(const h2*)&As[m][kb];
                afrag[s][2 * j]     = p.x;
                afrag[s][2 * j + 1] = p.y;
            }
        }
        #pragma unroll
        for (int t = 0; t < 2; ++t) {
            int n = wn * 32 + t * 16 + mrow;
            #pragma unroll
            for (int j = 0; j < 8; ++j) {
                int kb = ((j < 4) ? (j * 2) : (16 + (j - 4) * 2)) + khalf * 8;
                h2 p = *(const h2*)&Bs[n][kb];
                bfrag[t][2 * j]     = p.x;
                bfrag[t][2 * j + 1] = p.y;
            }
        }

        #pragma unroll
        for (int s = 0; s < 4; ++s)
            #pragma unroll
            for (int t = 0; t < 2; ++t)
                acc[s][t] = __builtin_amdgcn_wmma_f32_16x16x32_f16(
                    false, afrag[s], false, bfrag[t],
                    (short)0, acc[s][t], false, false);
    }

    // store C fragments: VGPR r -> M = r + 8*(lane>>4), N = lane&15
    #pragma unroll
    for (int s = 0; s < 4; ++s)
        #pragma unroll
        for (int t = 0; t < 2; ++t)
            #pragma unroll
            for (int r = 0; r < 8; ++r) {
                int row = tileM + wm * 64 + s * 16 + khalf * 8 + r;
                int col = tileN + wn * 32 + t * 16 + mrow;
                Cout[(size_t)row * LD + col] = acc[s][t][r];
            }
}

// ---------------- 3) fill alpha bins (last row / col / corner) ----------------
__global__ void k_fill_bins(float* __restrict__ C, const float* __restrict__ alpha) {
    int idx = blockIdx.x * blockDim.x + threadIdx.x;
    float a = alpha[0];
    if (idx < LD)     C[(size_t)M_ROWS * LD + idx] = a;   // last row (incl corner)
    if (idx < M_ROWS) C[(size_t)idx * LD + N_COLS] = a;   // last col
}

__global__ void k_init_uv(float* __restrict__ ws) {
    int idx = blockIdx.x * blockDim.x + threadIdx.x;
    if (idx < LD) { ws[WS_U + idx] = 0.0f; ws[WS_V + idx] = 0.0f; }
}

// -------- 4a) u[i] = log_mu[i] - logsumexp_j(C[i,j] + v[j]) : block per row --------
__global__ void k_sinkhorn_u(const float* __restrict__ C, const float* __restrict__ v,
                             float* __restrict__ u) {
    const int i   = blockIdx.x;
    const int tid = threadIdx.x;
    const float NEG = -__builtin_inff();
    float m = NEG, s = 0.0f;
    const float* row = C + (size_t)i * LD;
    for (int j = tid; j < LD; j += 256) {
        lse_combine(m, s, row[j] + v[j], 1.0f);
    }
    for (int off = 16; off; off >>= 1) {
        float om = __shfl_xor(m, off, 32);
        float os = __shfl_xor(s, off, 32);
        lse_combine(m, s, om, os);
    }
    __shared__ float sm[8], ss[8];
    int wave = tid >> 5, lane = tid & 31;
    if (lane == 0) { sm[wave] = m; ss[wave] = s; }
    __syncthreads();
    if (tid == 0) {
        for (int w = 1; w < 8; ++w) lse_combine(m, s, sm[w], ss[w]);
        const float norm   = -__logf(12288.0f);
        const float log_mu = (i < M_ROWS) ? norm : (__logf((float)N_COLS) + norm);
        u[i] = log_mu - (m + __logf(s));
    }
}

// -------- 4b) column-wise partial logsumexp over row chunks (coalesced) --------
__global__ void k_sinkhorn_v_partial(const float* __restrict__ C, const float* __restrict__ u,
                                     float* __restrict__ pm, float* __restrict__ ps) {
    const int j     = blockIdx.x * blockDim.x + threadIdx.x;
    const int chunk = blockIdx.y;
    const int rpc   = (LD + NCHUNK - 1) / NCHUNK;
    int i0 = chunk * rpc;
    int i1 = i0 + rpc; if (i1 > LD) i1 = LD;
    if (j >= LD) return;
    const float NEG = -__builtin_inff();
    float m = NEG, s = 0.0f;
    for (int i = i0; i < i1; ++i) {
        lse_combine(m, s, C[(size_t)i * LD + j] + u[i], 1.0f);
    }
    pm[chunk * PSTRIDE + j] = m;
    ps[chunk * PSTRIDE + j] = s;
}

__global__ void k_sinkhorn_v_combine(const float* __restrict__ pm, const float* __restrict__ ps,
                                     float* __restrict__ v) {
    const int j = blockIdx.x * blockDim.x + threadIdx.x;
    if (j >= LD) return;
    const float NEG = -__builtin_inff();
    float m = NEG, s = 0.0f;
    for (int c = 0; c < NCHUNK; ++c)
        lse_combine(m, s, pm[c * PSTRIDE + j], ps[c * PSTRIDE + j]);
    const float norm   = -__logf(12288.0f);
    const float log_nu = (j < N_COLS) ? norm : (__logf((float)M_ROWS) + norm);
    v[j] = log_nu - (m + __logf(s));
}

// ---------------- 5) Z = C + u + v - norm, in place on d_out ----------------
__global__ void k_finalize(float* __restrict__ Z, const float* __restrict__ u,
                           const float* __restrict__ v) {
    const int j = blockIdx.x * blockDim.x + threadIdx.x;
    const int i = blockIdx.y;
    if (j >= LD) return;
    const float norm = -__logf(12288.0f);
    Z[(size_t)i * LD + j] += u[i] + v[j] - norm;
}

extern "C" void kernel_launch(void* const* d_in, const int* in_sizes, int n_in,
                              void* d_out, int out_size, void* d_ws, size_t ws_size,
                              hipStream_t stream) {
    const float* x     = (const float*)d_in[0];
    const float* alpha = (const float*)d_in[1];
    float* ws = (float*)d_ws;
    float* C  = (float*)d_out;          // couplings built in place, finalized to Z

    k_minmax_partial<<<256, 256, 0, stream>>>(x, ws);
    k_minmax_final<<<1, 256, 0, stream>>>(ws);

    dim3 gemm_grid(N_COLS / 128, M_ROWS / 128);     // 48 x 48
    k_gemm_wmma<<<gemm_grid, 256, 0, stream>>>(x, ws, C);

    const int nb = (LD + 255) / 256;                // 25
    k_fill_bins<<<nb, 256, 0, stream>>>(C, alpha);
    k_init_uv<<<nb, 256, 0, stream>>>(ws);

    for (int it = 0; it < 100; ++it) {
        k_sinkhorn_u<<<LD, 256, 0, stream>>>(C, ws + WS_V, ws + WS_U);
        dim3 vp_grid(nb, NCHUNK);
        k_sinkhorn_v_partial<<<vp_grid, 256, 0, stream>>>(C, ws + WS_U, ws + WS_PM, ws + WS_PS);
        k_sinkhorn_v_combine<<<nb, 256, 0, stream>>>(ws + WS_PM, ws + WS_PS, ws + WS_V);
    }

    dim3 fin_grid(nb, LD);
    k_finalize<<<fin_grid, 256, 0, stream>>>(C, ws + WS_U, ws + WS_V);
}
